// SelfAttentionLayer_71262097376070
// MI455X (gfx1250) — matrix operand order repaired and checked by reading, not compiled, not run on hardware
//
#include <hip/hip_runtime.h>

// ---------------------------------------------------------------------------
// MI455X (gfx1250) implementation of the reference "self-attention layer".
// Heavy math: v_wmma_f32_16x16x32_bf16 (wave32 WMMA).
// Attention K-tiles staged with global_load_async_to_lds_b128 (ASYNCcnt path)
// and double-buffered so staging overlaps WMMA compute.
// ---------------------------------------------------------------------------

typedef __attribute__((ext_vector_type(16))) __bf16        v16bf;
typedef __attribute__((ext_vector_type(8)))  float         v8f;
typedef __attribute__((ext_vector_type(4)))  float         vf4;

union FragU { vf4 f4[2]; v16bf v; };

__device__ __forceinline__ __bf16 f2bf(float f) {
  unsigned int u = __float_as_uint(f);
  unsigned int r = (u + 0x7FFFu + ((u >> 16) & 1u)) >> 16;   // round-nearest-even
  unsigned short s = (unsigned short)r;
  __bf16 b;
  __builtin_memcpy(&b, &s, 2);
  return b;
}

// A-matrix fragment (16x32, M x K) from row-major bf16 LDS with `stride`.
// Lane L (and L+16) holds row M = m0 + (L&15); K = base..base+7, base+16..base+23,
// base = 8*(L>=16).  (CDNA5 ISA 7.12.2, 16-bit A layout)
__device__ __forceinline__ v16bf frag_A(const __bf16* lds, int stride, int m0, int kk) {
  const int lane = threadIdx.x & 31;
  const int m  = m0 + (lane & 15);
  const int kb = kk + ((lane >> 4) << 3);
  const __bf16* p = lds + m * stride + kb;
  FragU fr;
  fr.f4[0] = *(const vf4*)(p);
  fr.f4[1] = *(const vf4*)(p + 16);
  return fr.v;
}

// B-matrix fragment (32x16, K x N) from row-major [n][k] bf16 LDS.
// Lane L holds column N = n0 + (L&15); K = base..base+7, base+16..base+23, base = 8*(L>=16).
__device__ __forceinline__ v16bf frag_B(const __bf16* lds, int stride, int n0, int kk) {
  const int lane = threadIdx.x & 31;
  const int n  = n0 + (lane & 15);
  const int kb = kk + ((lane >> 4) << 3);
  const __bf16* p = lds + n * stride + kb;
  FragU fr;
  fr.f4[0] = *(const vf4*)(p);
  fr.f4[1] = *(const vf4*)(p + 16);
  return fr.v;
}

__device__ __forceinline__ v8f wmma_bf16(v16bf a, v16bf b, v8f c) {
  return __builtin_amdgcn_wmma_f32_16x16x32_bf16(false, a, false, b, (short)0, c,
                                                 false, false);
}

// ---------------------------------------------------------------------------
// Async global -> LDS staging of ROWS x 512 bf16 (row stride in LDS = 520).
// NB = b128 transfers per thread; 128 threads cover ROWS = NB*2 rows.
// IOFFSET applies to both the LDS and global addresses (ISA 15.18 pseudocode),
// so a single base per thread + immediate offsets suffices.
// ---------------------------------------------------------------------------
template <int NB>
__device__ __forceinline__ void async_stage(const __bf16* g_base, __bf16* lds_base) {
  const int t   = threadIdx.x;
  const int tpr = 64 / NB;                   // threads per row (64 b128 per row)
  const int row = t / tpr;
  const int col = (t % tpr) * NB * 8;        // element offset within row
  const unsigned long long g =
      (unsigned long long)(size_t)(const void*)(g_base + (size_t)row * 512 + col);
  const unsigned int l =
      (unsigned int)(size_t)(void*)(lds_base + row * 520 + col);
#pragma unroll
  for (int i = 0; i < NB; ++i) {
    asm volatile("global_load_async_to_lds_b128 %0, %1, off offset:%c2"
                 :: "v"(l), "v"(g), "i"(i * 16) : "memory");
  }
}

__device__ __forceinline__ void wait_async_le16() {
  asm volatile("s_wait_asynccnt 16" ::: "memory");
}
__device__ __forceinline__ void wait_async_0() {
  asm volatile("s_wait_asynccnt 0" ::: "memory");
}

// ---------------------------------------------------------------------------
// Projection GEMM:  O[8192 x 512] = A[8192 x 512] @ W[512 x 512] + bias
// 128 threads (4 waves), tile 64(M) x 64(N), K-step 32.
// ---------------------------------------------------------------------------
template <bool OUT_BF16>
__global__ __launch_bounds__(128)
void gemm_proj(const float* __restrict__ A, const float* __restrict__ W,
               const float* __restrict__ bias,
               __bf16* __restrict__ Obf, float* __restrict__ Of)
{
  constexpr int LDT = 40;                       // 32 + 8 pad (keeps 16B align)
  __shared__ __bf16 sA[64 * LDT];
  __shared__ __bf16 sB[64 * LDT];
  const int t = threadIdx.x, wave = t >> 5, lane = t & 31;
  const int m0 = blockIdx.x * 64;
  const int n0 = blockIdx.y * 64;

  const v8f vz = {0.f, 0.f, 0.f, 0.f, 0.f, 0.f, 0.f, 0.f};
  v8f acc[4] = {vz, vz, vz, vz};

  for (int kk = 0; kk < 512; kk += 32) {
    __syncthreads();
    { // A tile 64x32 (fp32 -> bf16)
      const int row = t >> 1, seg = (t & 1) * 16;
      const vf4* g = (const vf4*)(A + ((size_t)(m0 + row)) * 512 + kk + seg);
      __bf16* d = sA + row * LDT + seg;
#pragma unroll
      for (int i = 0; i < 4; ++i) {
        vf4 f = g[i];
        d[4*i+0] = f2bf(f.x); d[4*i+1] = f2bf(f.y);
        d[4*i+2] = f2bf(f.z); d[4*i+3] = f2bf(f.w);
      }
    }
    { // W tile 32x64 -> sB[n][k] (transposed in LDS)
      const int krow = t >> 2, nseg = (t & 3) * 16;
      const vf4* g = (const vf4*)(W + ((size_t)(kk + krow)) * 512 + n0 + nseg);
#pragma unroll
      for (int i = 0; i < 4; ++i) {
        vf4 f = g[i];
        sB[(nseg + 4*i + 0) * LDT + krow] = f2bf(f.x);
        sB[(nseg + 4*i + 1) * LDT + krow] = f2bf(f.y);
        sB[(nseg + 4*i + 2) * LDT + krow] = f2bf(f.z);
        sB[(nseg + 4*i + 3) * LDT + krow] = f2bf(f.w);
      }
    }
    __syncthreads();
    v16bf a = frag_A(sA, LDT, wave * 16, 0);
#pragma unroll
    for (int nt = 0; nt < 4; ++nt) {
      v16bf b = frag_B(sB, LDT, nt * 16, 0);
      acc[nt] = wmma_bf16(a, b, acc[nt]);
    }
  }

  // Epilogue (C/D layout: lane -> N, vgpr j -> M = j + 8*(lane>=16))
  const int mrow = m0 + wave * 16 + ((lane >> 4) << 3);
#pragma unroll
  for (int nt = 0; nt < 4; ++nt) {
    const int n = n0 + nt * 16 + (lane & 15);
    const float bn_ = bias[n];
#pragma unroll
    for (int j = 0; j < 8; ++j) {
      const float v = acc[nt][j] + bn_;
      const size_t o = ((size_t)(mrow + j)) * 512 + n;
      if (OUT_BF16) Obf[o] = f2bf(v); else Of[o] = v;
    }
  }
}

// ---------------------------------------------------------------------------
// Attention pass 1: per-row max / sum-exp of S = Q K^T (no 1/sqrt(d) in ref!).
// Computed as S^T tiles (A = K rows, B = Q^T) so row reductions are in-lane.
// K tiles (32 rows) double-buffered via async-to-LDS.
// Grid: (64 m-blocks, 4 n-chunks of 1024, 2 batches); 128 threads.
// ---------------------------------------------------------------------------
__global__ __launch_bounds__(128)
void attn_rowstats(const __bf16* __restrict__ Q, const __bf16* __restrict__ K,
                   float* __restrict__ pmax, float* __restrict__ psum)
{
  constexpr int LDQ = 520;                      // 512 + 8 pad
  extern __shared__ __bf16 smem[];
  __bf16* sQ  = smem;                           // 64 x 520 (this WG's m rows)
  __bf16* sK0 = smem + 64 * LDQ;                // 32 x 520 (K tile, buffer 0)
  __bf16* sK1 = sK0 + 32 * LDQ;                 // 32 x 520 (K tile, buffer 1)
  const int t = threadIdx.x, wave = t >> 5, lane = t & 31;
  const int b   = blockIdx.z;
  const int m0  = blockIdx.x * 64;
  const int nc0 = blockIdx.y * 1024;
  const __bf16* Qg = Q + ((size_t)b * 4096 + m0) * 512;
  const __bf16* Kg = K + ((size_t)b * 4096 + nc0) * 512;

  async_stage<32>(Qg, sQ);                      // Q block (64 rows)
  async_stage<16>(Kg, sK0);                     // K tile 0 (32 rows)

  float rmax = -__builtin_inff(), rsum = 0.f;
  const v8f vz = {0.f, 0.f, 0.f, 0.f, 0.f, 0.f, 0.f, 0.f};

  for (int st = 0; st < 32; ++st) {
    __bf16* cur = (st & 1) ? sK1 : sK0;
    __bf16* nxt = (st & 1) ? sK0 : sK1;
    if (st + 1 < 32) {                          // prefetch next K tile
      async_stage<16>(Kg + (size_t)(st + 1) * 32 * 512, nxt);
      wait_async_le16();                        // tile st complete; st+1 in flight
    } else {
      wait_async_0();
    }
    __syncthreads();                            // tile st visible to all waves

    v8f acc[2] = {vz, vz};
#pragma unroll 4
    for (int kk = 0; kk < 512; kk += 32) {
      v16bf bq = frag_B(sQ, LDQ, wave * 16, kk);   // B = Q^T, lane -> m
#pragma unroll
      for (int nt = 0; nt < 2; ++nt) {
        v16bf a = frag_A(cur, LDQ, nt * 16, kk);   // A = K rows, lane -> n
        acc[nt] = wmma_bf16(a, bq, acc[nt]);
      }
    }
    // online softmax stats: this lane sees 16 n-values of a fixed row m
    float tmax = -__builtin_inff();
#pragma unroll
    for (int nt = 0; nt < 2; ++nt)
#pragma unroll
      for (int j = 0; j < 8; ++j) tmax = fmaxf(tmax, acc[nt][j]);
    const float nm = fmaxf(rmax, tmax);
    rsum *= __expf(rmax - nm);
#pragma unroll
    for (int nt = 0; nt < 2; ++nt)
#pragma unroll
      for (int j = 0; j < 8; ++j) rsum += __expf(acc[nt][j] - nm);
    rmax = nm;
    __syncthreads();                            // all waves done reading `cur`
  }

  // lanes L and L+16 hold the same m over disjoint n halves: merge.
  const float omax = __shfl_xor(rmax, 16, 32);
  const float osum = __shfl_xor(rsum, 16, 32);
  const float gm = fmaxf(rmax, omax);
  const float gs = rsum * __expf(rmax - gm) + osum * __expf(omax - gm);
  if (lane < 16) {
    const int m = m0 + wave * 16 + lane;
    const size_t idx = ((size_t)b * 4096 + m) * 4 + blockIdx.y;
    pmax[idx] = gm;
    psum[idx] = gs;
  }
}

__global__ void rowstats_combine(const float* __restrict__ pmax,
                                 const float* __restrict__ psum,
                                 float* __restrict__ rowmax,
                                 float* __restrict__ rowsuminv)
{
  const int i = blockIdx.x * blockDim.x + threadIdx.x;
  if (i >= 2 * 4096) return;
  float gm = -__builtin_inff();
#pragma unroll
  for (int c = 0; c < 4; ++c) gm = fmaxf(gm, pmax[i * 4 + c]);
  float gs = 0.f;
#pragma unroll
  for (int c = 0; c < 4; ++c) gs += psum[i * 4 + c] * __expf(pmax[i * 4 + c] - gm);
  rowmax[i] = gm;
  rowsuminv[i] = 1.0f / gs;
}

// ---------------------------------------------------------------------------
// Attention pass 2: colsum[n] += sum_m exp(S[m,n] - rowmax[m]) * rowsuminv[m].
// Standard orientation (A = Q rows, B = K^T); m-reduction in-lane.
// Same async double-buffered K tiles.
// ---------------------------------------------------------------------------
__global__ __launch_bounds__(128)
void attn_colsum(const __bf16* __restrict__ Q, const __bf16* __restrict__ K,
                 const float* __restrict__ rowmax, const float* __restrict__ rowsuminv,
                 float* __restrict__ colsum)
{
  constexpr int LDQ = 520;
  extern __shared__ __bf16 smem[];
  __bf16* sQ  = smem;
  __bf16* sK0 = smem + 64 * LDQ;
  __bf16* sK1 = sK0 + 32 * LDQ;
  const int t = threadIdx.x, wave = t >> 5, lane = t & 31;
  const int b   = blockIdx.z;
  const int m0  = blockIdx.x * 64;
  const int nc0 = blockIdx.y * 1024;
  const __bf16* Qg = Q + ((size_t)b * 4096 + m0) * 512;
  const __bf16* Kg = K + ((size_t)b * 4096 + nc0) * 512;

  // row stats for this lane's 8 m values (D layout: m = j + 8*(lane>=16))
  float rm[8], ri[8];
  {
    const int mb = b * 4096 + m0 + wave * 16 + ((lane >> 4) << 3);
#pragma unroll
    for (int j = 0; j < 8; ++j) { rm[j] = rowmax[mb + j]; ri[j] = rowsuminv[mb + j]; }
  }

  async_stage<32>(Qg, sQ);
  async_stage<16>(Kg, sK0);

  const v8f vz = {0.f, 0.f, 0.f, 0.f, 0.f, 0.f, 0.f, 0.f};

  for (int st = 0; st < 32; ++st) {
    __bf16* cur = (st & 1) ? sK1 : sK0;
    __bf16* nxt = (st & 1) ? sK0 : sK1;
    if (st + 1 < 32) {
      async_stage<16>(Kg + (size_t)(st + 1) * 32 * 512, nxt);
      wait_async_le16();
    } else {
      wait_async_0();
    }
    __syncthreads();

    v8f acc[2] = {vz, vz};
#pragma unroll 4
    for (int kk = 0; kk < 512; kk += 32) {
      v16bf a = frag_A(sQ, LDQ, wave * 16, kk);    // A = Q rows (lane -> m)
#pragma unroll
      for (int nt = 0; nt < 2; ++nt) {
        v16bf bk = frag_B(cur, LDQ, nt * 16, kk);  // B = K^T (lane -> n)
        acc[nt] = wmma_bf16(a, bk, acc[nt]);
      }
    }
#pragma unroll
    for (int nt = 0; nt < 2; ++nt) {
      float part = 0.f;
#pragma unroll
      for (int j = 0; j < 8; ++j) part += __expf(acc[nt][j] - rm[j]) * ri[j];
      part += __shfl_xor(part, 16, 32);            // merge m-halves (same n)
      if (lane < 16)
        atomicAdd(&colsum[(size_t)b * 4096 + nc0 + st * 32 + nt * 16 + lane], part);
    }
    __syncthreads();
  }
}

// ---------------------------------------------------------------------------
// Output GEMM + fused epilogue:
//   out = relu( BN( (V * colsum/(eps+colsum)) @ Wo + bo ) ) + inputs
// ---------------------------------------------------------------------------
__global__ __launch_bounds__(128)
void gemm_out(const float* __restrict__ V, const float* __restrict__ colsum,
              const float* __restrict__ Wo, const float* __restrict__ bo,
              const float* __restrict__ gamma, const float* __restrict__ beta,
              const float* __restrict__ mean, const float* __restrict__ var,
              const float* __restrict__ inputs, float* __restrict__ out)
{
  constexpr int LDT = 40;
  __shared__ __bf16 sA[64 * LDT];
  __shared__ __bf16 sB[64 * LDT];
  const int t = threadIdx.x, wave = t >> 5, lane = t & 31;
  const int m0 = blockIdx.x * 64;
  const int n0 = blockIdx.y * 64;

  const v8f vz = {0.f, 0.f, 0.f, 0.f, 0.f, 0.f, 0.f, 0.f};
  v8f acc[4] = {vz, vz, vz, vz};

  for (int kk = 0; kk < 512; kk += 32) {
    __syncthreads();
    { // A tile = V * scale(row), fp32 -> bf16
      const int row = t >> 1, seg = (t & 1) * 16;
      const float cs = colsum[m0 + row];
      const float sc = cs / (1e-9f + cs);
      const vf4* g = (const vf4*)(V + ((size_t)(m0 + row)) * 512 + kk + seg);
      __bf16* d = sA + row * LDT + seg;
#pragma unroll
      for (int i = 0; i < 4; ++i) {
        vf4 f = g[i];
        d[4*i+0] = f2bf(f.x * sc); d[4*i+1] = f2bf(f.y * sc);
        d[4*i+2] = f2bf(f.z * sc); d[4*i+3] = f2bf(f.w * sc);
      }
    }
    {
      const int krow = t >> 2, nseg = (t & 3) * 16;
      const vf4* g = (const vf4*)(Wo + ((size_t)(kk + krow)) * 512 + n0 + nseg);
#pragma unroll
      for (int i = 0; i < 4; ++i) {
        vf4 f = g[i];
        sB[(nseg + 4*i + 0) * LDT + krow] = f2bf(f.x);
        sB[(nseg + 4*i + 1) * LDT + krow] = f2bf(f.y);
        sB[(nseg + 4*i + 2) * LDT + krow] = f2bf(f.z);
        sB[(nseg + 4*i + 3) * LDT + krow] = f2bf(f.w);
      }
    }
    __syncthreads();
    v16bf a = frag_A(sA, LDT, wave * 16, 0);
#pragma unroll
    for (int nt = 0; nt < 4; ++nt) {
      v16bf b = frag_B(sB, LDT, nt * 16, 0);
      acc[nt] = wmma_bf16(a, b, acc[nt]);
    }
  }

  const int mrow = m0 + wave * 16 + ((lane >> 4) << 3);
#pragma unroll
  for (int nt = 0; nt < 4; ++nt) {
    const int n = n0 + nt * 16 + (lane & 15);
    const float bb = bo[n];
    const float mu = mean[n];
    const float iv = rsqrtf(var[n] + 1e-5f);
    const float gm = gamma[n];
    const float bt = beta[n];
#pragma unroll
    for (int j = 0; j < 8; ++j) {
      float v = acc[nt][j] + bb;
      v = (v - mu) * iv * gm + bt;
      v = fmaxf(v, 0.f);
      const size_t o = ((size_t)(mrow + j)) * 512 + n;
      out[o] = v + inputs[o];
    }
  }
}

// ---------------------------------------------------------------------------
// Host-side launch
// ---------------------------------------------------------------------------
extern "C" void kernel_launch(void* const* d_in, const int* in_sizes, int n_in,
                              void* d_out, int out_size, void* d_ws, size_t ws_size,
                              hipStream_t stream)
{
  (void)in_sizes; (void)n_in; (void)out_size; (void)ws_size;
  const float* inputs  = (const float*)d_in[0];
  const float* Wq      = (const float*)d_in[1];
  const float* Wk      = (const float*)d_in[2];
  const float* Wv      = (const float*)d_in[3];
  const float* Wo      = (const float*)d_in[4];
  const float* bq      = (const float*)d_in[5];
  const float* bk      = (const float*)d_in[6];
  const float* bv      = (const float*)d_in[7];
  const float* bo      = (const float*)d_in[8];
  const float* bnscale = (const float*)d_in[9];
  const float* bnbias  = (const float*)d_in[10];
  const float* bnmean  = (const float*)d_in[11];
  const float* bnvar   = (const float*)d_in[12];
  float* out = (float*)d_out;

  char* ws = (char*)d_ws;
  __bf16* Qb  = (__bf16*)(ws);                               //  8 MiB
  __bf16* Kb  = (__bf16*)(ws + (size_t)(8u  << 20));         //  8 MiB
  float*  Vf  = (float*) (ws + (size_t)(16u << 20));         // 16 MiB
  float*  pmax      = (float*)(ws + (size_t)(32u << 20));            // 128 KiB
  float*  psum      = (float*)(ws + (size_t)(32u << 20) + 131072);   // 128 KiB
  float*  rowmax    = (float*)(ws + (size_t)(32u << 20) + 262144);   //  32 KiB
  float*  rowsuminv = (float*)(ws + (size_t)(32u << 20) + 294912);   //  32 KiB
  float*  colsumv   = (float*)(ws + (size_t)(32u << 20) + 327680);   //  32 KiB

  // colsum is accumulated with atomics -> must be zeroed every call.
  hipMemsetAsync(colsumv, 0, 2 * 4096 * sizeof(float), stream);

  const dim3 gP(128, 8, 1);   // 8192/64 x 512/64
  gemm_proj<true ><<<gP, 128, 0, stream>>>(inputs, Wq, bq, Qb, nullptr);
  gemm_proj<true ><<<gP, 128, 0, stream>>>(inputs, Wk, bk, Kb, nullptr);
  gemm_proj<false><<<gP, 128, 0, stream>>>(inputs, Wv, bv, nullptr, Vf);

  const size_t shm = (size_t)(64 + 32 + 32) * 520 * sizeof(__bf16);  // 130 KiB
  const dim3 gA(64, 4, 2);    // m-blocks x n-chunks x batch
  attn_rowstats<<<gA, 128, shm, stream>>>(Qb, Kb, pmax, psum);
  rowstats_combine<<<32, 256, 0, stream>>>(pmax, psum, rowmax, rowsuminv);
  attn_colsum<<<gA, 128, shm, stream>>>(Qb, Kb, rowmax, rowsuminv, colsumv);

  gemm_out<<<gP, 128, 0, stream>>>(Vf, colsumv, Wo, bo, bnscale, bnbias,
                                   bnmean, bnvar, inputs, out);
}
